// Head_37443524886846
// MI455X (gfx1250) — matrix operand order repaired and checked by reading, not compiled, not run on hardware
//
#include <hip/hip_runtime.h>

#define EMB 1024
#define HS 64
#define CTX 2048
#define NB 8

typedef __attribute__((ext_vector_type(16))) __bf16 v16bf;
typedef __attribute__((ext_vector_type(8)))  float  v8f;
typedef __attribute__((ext_vector_type(4)))  unsigned v4u;
typedef __attribute__((ext_vector_type(8)))  int      v8i;
typedef __attribute__((ext_vector_type(4)))  int      v4i;

union V16 {
    v16bf v;
    unsigned u[8];
};

__device__ __forceinline__ unsigned short f32_to_bf16(float f) {
    unsigned u = __builtin_bit_cast(unsigned, f);
    u += 0x7FFFu + ((u >> 16) & 1u);   // round-to-nearest-even
    return (unsigned short)(u >> 16);
}

// A-matrix 16x32 bf16 layout: VGPR j, half = lane>>4 -> k-pair base (contiguous k = 16*half + 2j)
__device__ __forceinline__ int kpatA(int j, int half) {
    return half * 16 + j * 2;
}

// ---------------------------------------------------------------------------
// TDM 2-D tile load: global -> LDS.  Builds D# groups per CDNA5 ISA ch.8.
//   data_size_code: 0=1B 1=2B 2=4B; dims/strides in elements.
//   tile1 rows of tile0 contiguous elements (rows stride0 apart in memory)
//   land densely at lds_byte_off.
// ---------------------------------------------------------------------------
__device__ __forceinline__ void tdm_load_2d(
        unsigned lds_byte_off, const void* gptr, unsigned data_size_code,
        unsigned tile0, unsigned tile1, unsigned long long stride0) {
    unsigned long long ga = (unsigned long long)(size_t)gptr;
    v4u g0;
    g0[0] = 1u;                                    // count=1 (valid), user bits 0
    g0[1] = lds_byte_off;                          // [63:32] lds_addr
    g0[2] = (unsigned)(ga & 0xffffffffu);          // [95:64] global_addr lo
    g0[3] = (unsigned)((ga >> 32) & 0x01ffffffu)   // [120:96] global_addr hi
          | 0x80000000u;                           // [127:126] type=2
    unsigned long long q0 = ((unsigned long long)data_size_code << 16)
                          | ((unsigned long long)(tile0 & 0xffffu) << 48);       // tensor_dim0 lo16
    unsigned long long q1 = ((unsigned long long)(tile0 >> 16))                  // tensor_dim0 hi16
                          | ((unsigned long long)tile1 << 16)                    // tensor_dim1
                          | ((unsigned long long)tile0 << 48);                   // tile_dim0
    unsigned long long q2 = ((unsigned long long)tile1)                          // tile_dim1
                          | ((stride0 & 0xffffffffull) << 32);                   // stride0[31:0]
    unsigned long long q3 = ((stride0 >> 32) & 0xffffull);                       // stride0[47:32]
    v8i g1;
    g1[0] = (int)(unsigned)(q0 & 0xffffffffull); g1[1] = (int)(unsigned)(q0 >> 32);
    g1[2] = (int)(unsigned)(q1 & 0xffffffffull); g1[3] = (int)(unsigned)(q1 >> 32);
    g1[4] = (int)(unsigned)(q2 & 0xffffffffull); g1[5] = (int)(unsigned)(q2 >> 32);
    g1[6] = (int)(unsigned)(q3 & 0xffffffffull); g1[7] = (int)(unsigned)(q3 >> 32);
    v4i g2 = {0, 0, 0, 0};
    v4i g3 = {0, 0, 0, 0};
#if defined(__clang_major__) && (__clang_major__ >= 23)
    v8i g4 = {0, 0, 0, 0, 0, 0, 0, 0};
    __builtin_amdgcn_tensor_load_to_lds(g0, g1, g2, g3, g4, 0);
#else
    __builtin_amdgcn_tensor_load_to_lds(g0, g1, g2, g3, 0);
#endif
}

// ---------------------------------------------------------------------------
// Kernel 0: W [E,H] f32  ->  Wt [3][H][E] bf16 (transposed, contiguous k-pairs)
// ---------------------------------------------------------------------------
__global__ void head_wt_kernel(const float* __restrict__ Wq,
                               const float* __restrict__ Wk,
                               const float* __restrict__ Wv,
                               unsigned short* __restrict__ Wt) {
    int idx = blockIdx.x * blockDim.x + threadIdx.x;
    if (idx >= 3 * HS * EMB) return;
    int mat = idx >> 16;          // HS*EMB = 65536
    int rem = idx & 65535;
    int h = rem >> 10;
    int e = rem & 1023;
    const float* W = (mat == 0) ? Wq : (mat == 1 ? Wk : Wv);
    Wt[idx] = f32_to_bf16(W[e * HS + h]);
}

// ---------------------------------------------------------------------------
// Kernel 1: QKV projection.  grid = 16384/64 blocks, 128 threads (4 waves).
// Both the X slab [64 rows x 64 e] (f32) and the Wt slab [192 rows x 64 e]
// (bf16) are TDM-DMA'd into double-buffered LDS; slab s+1 DMA overlaps the
// 24 bf16 WMMAs on slab s.  V is written transposed [B][64][CTX].
// ---------------------------------------------------------------------------
#define KSLAB 64
#define NSLAB (EMB / KSLAB)

__global__ __launch_bounds__(128) void head_qkv_kernel(
        const float* __restrict__ X, const unsigned short* __restrict__ Wt,
        unsigned short* __restrict__ Qo, unsigned short* __restrict__ Ko,
        unsigned short* __restrict__ Vt) {
    __shared__ float          ldsX[2][64 * KSLAB];    // 2 x 16 KB
    __shared__ unsigned short ldsW[2][192 * KSLAB];   // 2 x 24 KB

    const int lane = threadIdx.x & 31;
    const int wave = threadIdx.x >> 5;
    const int half = lane >> 4;
    const int m    = lane & 15;
    const int rowblk  = blockIdx.x * 64;
    const int rowbase = rowblk + wave * 16;
    const unsigned ldsX_off = (unsigned)(size_t)(void*)&ldsX[0][0];
    const unsigned ldsW_off = (unsigned)(size_t)(void*)&ldsW[0][0];

    v8f zero = {};
    v8f acc[3][4];
#pragma unroll
    for (int mt = 0; mt < 3; mt++)
#pragma unroll
        for (int nt = 0; nt < 4; nt++) acc[mt][nt] = zero;

    if (wave == 0) {                               // prologue: DMA slab 0
        tdm_load_2d(ldsX_off, X + (size_t)rowblk * EMB, 2, KSLAB, 64, EMB);
        tdm_load_2d(ldsW_off, Wt, 1, KSLAB, 192, EMB);
    }

    for (int st = 0; st < NSLAB; st++) {
        if (wave == 0) {
            if (st + 1 < NSLAB) {                  // DMA slab st+1 into other buffer
                int nb2 = (st + 1) & 1;
                tdm_load_2d(ldsX_off + nb2 * (64 * KSLAB * 4),
                            X + (size_t)rowblk * EMB + (st + 1) * KSLAB,
                            2, KSLAB, 64, EMB);
                tdm_load_2d(ldsW_off + nb2 * (192 * KSLAB * 2),
                            Wt + (st + 1) * KSLAB, 1, KSLAB, 192, EMB);
                __builtin_amdgcn_s_wait_tensorcnt(2);   // slab st complete (in-order)
            } else {
                __builtin_amdgcn_s_wait_tensorcnt(0);
            }
        }
        __syncthreads();                           // slab st visible to all

        const float*          xt = ldsX[st & 1];
        const unsigned short* wt = ldsW[st & 1];
#pragma unroll
        for (int ksub = 0; ksub < KSLAB / 32; ksub++) {
            V16 a;
#pragma unroll
            for (int j = 0; j < 8; j++) {
                int k = ksub * 32 + kpatA(j, half);
                float f0 = xt[(wave * 16 + m) * KSLAB + k];
                float f1 = xt[(wave * 16 + m) * KSLAB + k + 1];
                a.u[j] = (unsigned)f32_to_bf16(f0) | ((unsigned)f32_to_bf16(f1) << 16);
            }
#pragma unroll
            for (int mt = 0; mt < 3; mt++) {
#pragma unroll
                for (int nt = 0; nt < 4; nt++) {
                    V16 bm;
                    const unsigned short* wp =
                        wt + (mt * HS + nt * 16 + m) * KSLAB + ksub * 32 + half * 16;
#pragma unroll
                    for (int j = 0; j < 8; j++)
                        bm.u[j] = *(const unsigned*)(wp + 2 * j);
                    acc[mt][nt] = __builtin_amdgcn_wmma_f32_16x16x32_bf16(
                        false, a.v, false, bm.v, (short)0, acc[mt][nt], false, false);
                }
            }
        }
        __syncthreads();                           // all reads of slab st done
    }

#pragma unroll
    for (int nt = 0; nt < 4; nt++)
#pragma unroll
        for (int g = 0; g < 8; g++) {
            int r = rowbase + g + 8 * half;
            Qo[(size_t)r * HS + nt * 16 + m] = f32_to_bf16(acc[0][nt][g]);
            Ko[(size_t)r * HS + nt * 16 + m] = f32_to_bf16(acc[1][nt][g]);
            int bb = r >> 11;                      // batch
            int t  = r & (CTX - 1);
            Vt[((size_t)bb * HS + nt * 16 + m) * CTX + t] = f32_to_bf16(acc[2][nt][g]);
        }
}

// ---------------------------------------------------------------------------
// Kernel 2: fused causal flash attention, 64-key tiles.
// grid = NB * CTX/64 = 256 blocks, 128 threads (4 waves, 16 q-rows each).
// K [64 key][64 h] and V^T [64 h][64 key] tiles DMA'd by TDM, double buffered;
// the DMA of tile j+1 overlaps 16 WMMAs + online softmax on tile j.
// ---------------------------------------------------------------------------
__global__ __launch_bounds__(128) void head_attn_kernel(
        const unsigned short* __restrict__ Qb, const unsigned short* __restrict__ Kb,
        const unsigned short* __restrict__ Vt, float* __restrict__ out) {
    __shared__ unsigned short ldsK[2][64 * 64];    // [key][h]   2 x 8 KB
    __shared__ unsigned short ldsVT[2][64 * 64];   // [h][key]   2 x 8 KB
    __shared__ unsigned short ldsP[4][16 * 64];    // per-wave P [m][key] 8 KB

    const int lane = threadIdx.x & 31;
    const int wave = threadIdx.x >> 5;
    const int half = lane >> 4;
    const int m    = lane & 15;
    const int b    = blockIdx.x >> 5;              // CTX/64 = 32 q-blocks per batch
    const int q0   = (blockIdx.x & 31) * 64;
    const int qrow = q0 + wave * 16;
    const unsigned ldsK_off  = (unsigned)(size_t)(void*)&ldsK[0][0];
    const unsigned ldsVT_off = (unsigned)(size_t)(void*)&ldsVT[0][0];

    // Q A-tiles for this wave (loop invariant): two k-chunks over h=64
    V16 qa[2];
    {
        const unsigned short* qp = Qb + (size_t)(b * CTX + qrow + m) * HS;
#pragma unroll
        for (int c = 0; c < 2; c++)
#pragma unroll
            for (int j = 0; j < 8; j++)
                qa[c].u[j] = *(const unsigned*)(qp + c * 32 + kpatA(j, half));
    }

    v8f zero = {};
    v8f o[4];
#pragma unroll
    for (int nt = 0; nt < 4; nt++) o[nt] = zero;
    float mrow[8], lrow[8];
#pragma unroll
    for (int g = 0; g < 8; g++) { mrow[g] = -1e30f; lrow[g] = 0.0f; }

    const float scale = 0.125f;                    // 1/sqrt(64)
    const int ntiles = q0 / 64 + 1;                // keys 0 .. q0+63

    if (wave == 0) {                               // prologue: DMA tile 0
        tdm_load_2d(ldsK_off,  Kb + (size_t)(b * CTX) * HS, 1, 64, 64, HS);
        tdm_load_2d(ldsVT_off, Vt + (size_t)(b * HS) * CTX, 1, 64, 64, CTX);
    }

    for (int jt = 0; jt < ntiles; jt++) {
        const int key0 = jt * 64;
        const int cur  = jt & 1;

        if (wave == 0) {
            if (jt + 1 < ntiles) {                 // DMA tile jt+1 into other buffer
                int nb2 = (jt + 1) & 1;
                tdm_load_2d(ldsK_off + nb2 * (64 * 64 * 2),
                            Kb + (size_t)(b * CTX + key0 + 64) * HS, 1, 64, 64, HS);
                tdm_load_2d(ldsVT_off + nb2 * (64 * 64 * 2),
                            Vt + (size_t)(b * HS) * CTX + key0 + 64, 1, 64, 64, CTX);
                __builtin_amdgcn_s_wait_tensorcnt(2);   // pair jt complete (in-order)
            } else {
                __builtin_amdgcn_s_wait_tensorcnt(0);
            }
        }
        __syncthreads();                           // barrier A: tile jt ready

        // S = Q K^T : four 16x16 key tiles, 2 k-chunks each
        const unsigned short* kt = ldsK[cur];
        v8f s[4];
#pragma unroll
        for (int nt2 = 0; nt2 < 4; nt2++) {
            s[nt2] = zero;
#pragma unroll
            for (int c = 0; c < 2; c++) {
                V16 bk;
                const unsigned short* kp = kt + (nt2 * 16 + m) * 64 + c * 32 + half * 16;
#pragma unroll
                for (int j = 0; j < 8; j++)
                    bk.u[j] = *(const unsigned*)(kp + 2 * j);
                s[nt2] = __builtin_amdgcn_wmma_f32_16x16x32_bf16(
                    false, qa[c].v, false, bk.v, (short)0, s[nt2], false, false);
            }
        }

        // scale + causal mask; element (row = qrow+g+8*half, key = key0+nt2*16+m)
#pragma unroll
        for (int nt2 = 0; nt2 < 4; nt2++)
#pragma unroll
            for (int g = 0; g < 8; g++) {
                int key = key0 + nt2 * 16 + m;
                int qr  = qrow + g + 8 * half;
                float v = s[nt2][g] * scale;
                s[nt2][g] = (key <= qr) ? v : -1e30f;
            }

        // online softmax: row reductions across 16-lane halves (wave32)
#pragma unroll
        for (int g = 0; g < 8; g++) {
            float t = fmaxf(fmaxf(s[0][g], s[1][g]), fmaxf(s[2][g], s[3][g]));
            t = fmaxf(t, __shfl_xor(t, 1));
            t = fmaxf(t, __shfl_xor(t, 2));
            t = fmaxf(t, __shfl_xor(t, 4));
            t = fmaxf(t, __shfl_xor(t, 8));
            float mnew = fmaxf(mrow[g], t);
            float corr = __expf(mrow[g] - mnew);
            mrow[g] = mnew;
            float rs = 0.0f;
#pragma unroll
            for (int nt2 = 0; nt2 < 4; nt2++) {
                float p = __expf(s[nt2][g] - mnew);
                s[nt2][g] = p;
                rs += p;
            }
            rs += __shfl_xor(rs, 1);
            rs += __shfl_xor(rs, 2);
            rs += __shfl_xor(rs, 4);
            rs += __shfl_xor(rs, 8);
            lrow[g] = lrow[g] * corr + rs;
#pragma unroll
            for (int nt = 0; nt < 4; nt++) o[nt][g] *= corr;
        }

        // P: C layout -> A layout via per-wave LDS scratch [16 m][64 key]
        unsigned short* pp = ldsP[wave];
#pragma unroll
        for (int nt2 = 0; nt2 < 4; nt2++)
#pragma unroll
            for (int g = 0; g < 8; g++)
                pp[(g + 8 * half) * 64 + nt2 * 16 + m] = f32_to_bf16(s[nt2][g]);
        __syncthreads();                           // barrier B: P ready

        // O += P @ V  (A = P 16x64 = 2 tiles, B = V^T tiles 64key x 16h)
        const unsigned short* vt = ldsVT[cur];
        V16 pa[2];
#pragma unroll
        for (int c2 = 0; c2 < 2; c2++)
#pragma unroll
            for (int j = 0; j < 8; j++)
                pa[c2].u[j] = *(const unsigned*)(pp + m * 64 + c2 * 32 + kpatA(j, half));
#pragma unroll
        for (int nt = 0; nt < 4; nt++) {
#pragma unroll
            for (int c2 = 0; c2 < 2; c2++) {
                V16 bv;
                const unsigned short* vp = vt + (nt * 16 + m) * 64 + c2 * 32 + half * 16;
#pragma unroll
                for (int j = 0; j < 8; j++)
                    bv.u[j] = *(const unsigned*)(vp + 2 * j);
                o[nt] = __builtin_amdgcn_wmma_f32_16x16x32_bf16(
                    false, pa[c2].v, false, bv.v, (short)0, o[nt], false, false);
            }
        }
        __syncthreads();                           // barrier C: buf cur fully read
    }

    // epilogue: out = O / l  (streaming, non-temporal)
#pragma unroll
    for (int nt = 0; nt < 4; nt++)
#pragma unroll
        for (int g = 0; g < 8; g++) {
            int r = qrow + g + 8 * half;
            __builtin_nontemporal_store(o[nt][g] / lrow[g],
                out + (size_t)(b * CTX + r) * HS + nt * 16 + m);
        }
}

// ---------------------------------------------------------------------------
extern "C" void kernel_launch(void* const* d_in, const int* in_sizes, int n_in,
                              void* d_out, int out_size, void* d_ws, size_t ws_size,
                              hipStream_t stream) {
    const float* X  = (const float*)d_in[0];
    const float* Wq = (const float*)d_in[1];
    const float* Wk = (const float*)d_in[2];
    const float* Wv = (const float*)d_in[3];
    float* out = (float*)d_out;

    unsigned short* Wt = (unsigned short*)d_ws;      // [3*64][1024] bf16
    unsigned short* Qb = Wt + 3 * HS * EMB;          // [B*T][64] bf16
    unsigned short* Kb = Qb + (size_t)NB * CTX * HS; // [B*T][64] bf16
    unsigned short* Vt = Kb + (size_t)NB * CTX * HS; // [B][64][CTX] bf16

    head_wt_kernel<<<(3 * HS * EMB + 255) / 256, 256, 0, stream>>>(Wq, Wk, Wv, Wt);
    head_qkv_kernel<<<(NB * CTX) / 64, 128, 0, stream>>>(X, Wt, Qb, Kb, Vt);
    head_attn_kernel<<<NB * (CTX / 64), 128, 0, stream>>>(Qb, Kb, Vt, out);
}
